// Discretization_34110630265312
// MI455X (gfx1250) — matrix operand order repaired and checked by reading, not compiled, not run on hardware
//
#include <hip/hip_runtime.h>
#include <hip/hip_bf16.h>

typedef __attribute__((ext_vector_type(16))) __bf16 v16bf;
typedef __attribute__((ext_vector_type(8)))  __bf16 v8bf;
typedef __attribute__((ext_vector_type(8)))  float  v8f;
typedef __attribute__((ext_vector_type(4)))  unsigned int v4u;
typedef __attribute__((ext_vector_type(8)))  int v8i;
typedef __attribute__((ext_vector_type(4)))  int v4i;

#define T_TOKENS 32768   // 512 * 64
#define DIM      256
#define KCODES   8192
#define WAVES_PB 8       // 256 threads = 8 wave32
#define ROWS_PB  (WAVES_PB * 16)   // 128 tokens per block
#define NTILES   (KCODES / 16)     // 512 code tiles
#define TILE_ELTS (16 * DIM)       // 4096 bf16 per plane
#define TILE_BF16 (2 * TILE_ELTS)  // hi+lo planes = 8192 bf16 = 16 KB

// ---------------------------------------------------------------------------
// Kernel 1: vhalf[k] = 0.5 * sum_d vocab[k][d]^2
// ---------------------------------------------------------------------------
__global__ void vq_v2half(const float* __restrict__ vocab,
                          float* __restrict__ vhalf) {
    int k = blockIdx.x * blockDim.x + threadIdx.x;
    if (k < KCODES) {
        const float4* p = reinterpret_cast<const float4*>(vocab + (size_t)k * DIM);
        float s = 0.f;
        #pragma unroll 8
        for (int i = 0; i < DIM / 4; ++i) {
            float4 f = p[i];
            s += f.x * f.x + f.y * f.y + f.z * f.z + f.w * f.w;
        }
        vhalf[k] = 0.5f * s;
    }
}

// ---------------------------------------------------------------------------
// Kernel 2: pre-convert vocab fp32 -> bf16 hi/lo planes, tile-major:
// wsB[tile][plane(hi=0,lo=1)][code-in-tile][d] -> contiguous 16 KB per tile.
// ---------------------------------------------------------------------------
__global__ void vq_prepB(const float* __restrict__ vocab,
                         __bf16* __restrict__ wsB) {
    const int tile = blockIdx.x;                // 512 blocks
    const int r  = threadIdx.x >> 4;            // code row in tile
    const int dB = (threadIdx.x & 15) * 16;     // 16 floats per thread
    const float* vp = vocab + ((size_t)tile * 16 + r) * DIM + dB;
    __bf16* hi = wsB + (size_t)tile * TILE_BF16 + r * DIM + dB;
    __bf16* lo = hi + TILE_ELTS;

    __bf16 th[16], tl[16];
    #pragma unroll
    for (int q = 0; q < 4; ++q) {
        float4 f = *reinterpret_cast<const float4*>(vp + q * 4);
        float vv[4] = {f.x, f.y, f.z, f.w};
        #pragma unroll
        for (int e = 0; e < 4; ++e) {
            __bf16 h = (__bf16)vv[e];
            th[q * 4 + e] = h;
            tl[q * 4 + e] = (__bf16)(vv[e] - (float)h);
        }
    }
    reinterpret_cast<v8bf*>(hi)[0] = *reinterpret_cast<v8bf*>(&th[0]);
    reinterpret_cast<v8bf*>(hi)[1] = *reinterpret_cast<v8bf*>(&th[8]);
    reinterpret_cast<v8bf*>(lo)[0] = *reinterpret_cast<v8bf*>(&tl[0]);
    reinterpret_cast<v8bf*>(lo)[1] = *reinterpret_cast<v8bf*>(&tl[8]);
}

// ---------------------------------------------------------------------------
// A-tile preload helper: 16x256 fp32 -> bf16 hi/lo register images in the
// ISA A layout (lane<16: K {0..7,16..23}; lane>=16: K {8..15,24..31}).
// ---------------------------------------------------------------------------
__device__ __forceinline__ void loadAtile(const float* __restrict__ seq,
                                          int aRow, int halfId,
                                          v16bf (&aH)[8], v16bf (&aL)[8]) {
    #pragma unroll
    for (int c = 0; c < 8; ++c) {
        const int d0 = c * 32 + halfId * 8;
        const float* xp = seq + (size_t)aRow * DIM + d0;
        float vals[16];
        float4 f;
        f = *reinterpret_cast<const float4*>(xp +  0);
        vals[0] = f.x; vals[1] = f.y; vals[2] = f.z; vals[3] = f.w;
        f = *reinterpret_cast<const float4*>(xp +  4);
        vals[4] = f.x; vals[5] = f.y; vals[6] = f.z; vals[7] = f.w;
        f = *reinterpret_cast<const float4*>(xp + 16);
        vals[8] = f.x; vals[9] = f.y; vals[10] = f.z; vals[11] = f.w;
        f = *reinterpret_cast<const float4*>(xp + 20);
        vals[12] = f.x; vals[13] = f.y; vals[14] = f.z; vals[15] = f.w;
        #pragma unroll
        for (int i = 0; i < 16; ++i) {
            __bf16 h = (__bf16)vals[i];
            aH[c][i] = h;
            aL[c][i] = (__bf16)(vals[i] - (float)h);
        }
    }
}

// ---------------------------------------------------------------------------
// Score tile + running-argmax helpers (shared by both main kernels)
// ---------------------------------------------------------------------------
__device__ __forceinline__ void scoreTile(const __bf16* ldsH, const __bf16* ldsL,
                                          const v16bf (&aH)[8], const v16bf (&aL)[8],
                                          int lane16, int halfId, float bias,
                                          int kBase, float (&bv)[8], int (&bi)[8]) {
    v8f acc0, acc1;
    #pragma unroll
    for (int j = 0; j < 8; ++j) { acc0[j] = bias; acc1[j] = 0.f; }

    #pragma unroll
    for (int c = 0; c < 8; ++c) {
        // B layout (32x16): lanes 0-15 K={0..7,16..23}, lanes 16-31 K={8..15,24..31}; N=lane&15.
        const int d0 = c * 32 + halfId * 8;
        v8bf bH0 = *reinterpret_cast<const v8bf*>(&ldsH[lane16 * DIM + d0]);
        v8bf bH1 = *reinterpret_cast<const v8bf*>(&ldsH[lane16 * DIM + d0 + 16]);
        v8bf bL0 = *reinterpret_cast<const v8bf*>(&ldsL[lane16 * DIM + d0]);
        v8bf bL1 = *reinterpret_cast<const v8bf*>(&ldsL[lane16 * DIM + d0 + 16]);
        v16bf bH = __builtin_shufflevector(bH0, bH1, 0,1,2,3,4,5,6,7,8,9,10,11,12,13,14,15);
        v16bf bL = __builtin_shufflevector(bL0, bL1, 0,1,2,3,4,5,6,7,8,9,10,11,12,13,14,15);
        acc0 = __builtin_amdgcn_wmma_f32_16x16x32_bf16(
            false, aH[c], false, bH, (short)0, acc0, false, false);
        acc1 = __builtin_amdgcn_wmma_f32_16x16x32_bf16(
            false, aH[c], false, bL, (short)0, acc1, false, false);
        acc1 = __builtin_amdgcn_wmma_f32_16x16x32_bf16(
            false, aL[c], false, bH, (short)0, acc1, false, false);
    }

    const int col = kBase + lane16;
    #pragma unroll
    for (int j = 0; j < 8; ++j) {
        float s = acc0[j] + acc1[j];
        if (s > bv[j]) { bv[j] = s; bi[j] = col; }
    }
}

__device__ __forceinline__ void reduceWrite(float (&bv)[8], int (&bi)[8],
                                            int lane16, int halfId, int rowBase,
                                            int* __restrict__ bestIdx) {
    #pragma unroll
    for (int j = 0; j < 8; ++j) {
        float v = bv[j];
        int   i = bi[j];
        #pragma unroll
        for (int off = 8; off >= 1; off >>= 1) {
            float ov = __shfl_xor(v, off, 32);
            int   oi = __shfl_xor(i, off, 32);
            if (ov > v || (ov == v && oi < i)) { v = ov; i = oi; }
        }
        if (lane16 == 0) bestIdx[rowBase + j + halfId * 8] = i;
    }
}

// ---------------------------------------------------------------------------
// Kernel 3 (TDM path): main argmin. Double-buffered LDS; wave 0 drives the
// Tensor Data Mover to pull pre-converted bf16 tiles; all 8 waves do WMMA.
// ---------------------------------------------------------------------------
__global__ void __launch_bounds__(256)
vq_argmin_wmma_tdm(const float* __restrict__ seq,
                   const __bf16* __restrict__ wsB,
                   const float* __restrict__ vhalf,
                   int* __restrict__ bestIdx) {
    __shared__ __align__(16) __bf16 lds[2][TILE_BF16];   // 2 x 16 KB

    const int tid    = threadIdx.x;
    const int lane   = tid & 31;
    const int wave   = tid >> 5;
    const int halfId = lane >> 4;
    const int lane16 = lane & 15;

    const int rowBase = (blockIdx.x * WAVES_PB + wave) * 16;
    v16bf aH[8], aL[8];
    loadAtile(seq, rowBase + lane16, halfId, aH, aL);

    float bv[8];
    int   bi[8];
    #pragma unroll
    for (int j = 0; j < 8; ++j) { bv[j] = -3.4e38f; bi[j] = 0; }

    const uint64_t gBase = (uint64_t)(uintptr_t)wsB;
    const unsigned ldsAddr0 = (unsigned)(uintptr_t)&lds[0][0];
    const unsigned ldsAddr1 = (unsigned)(uintptr_t)&lds[1][0];

    // Issue one 1-D TDM copy: 16 KB tile (hi+lo planes) -> LDS buffer.
    auto issueTDM = [&](int tile, int buf) {
        const uint64_t ga = gBase + (uint64_t)tile * (TILE_BF16 * 2); // bytes
        v4u g0;
        g0[0] = 0x1u;                                     // count=1, user mode
        g0[1] = (buf == 0) ? ldsAddr0 : ldsAddr1;         // lds_addr (bytes)
        g0[2] = (unsigned)ga;                             // global_addr[31:0]
        g0[3] = (unsigned)((ga >> 32) & 0x01FFFFFFu)      // global_addr[56:32]
                | (2u << 30);                             // type = 2 (image)
        v8i g1;
        g1[0] = 0x10000;                                  // data_size=1 (2 bytes)
        g1[1] = (int)((TILE_BF16 & 0xFFFFu) << 16);       // tensor_dim0[15:0]
        g1[2] = (int)(((TILE_BF16 >> 16) & 0xFFFFu)       // tensor_dim0[31:16]
                | (1u << 16));                            // tensor_dim1 = 1
        g1[3] = (int)((TILE_BF16 & 0xFFFFu) << 16);       // tile_dim0 = 8192
        g1[4] = 1;                                        // tile_dim1=1, tile_dim2=0
        g1[5] = TILE_BF16;                                // tensor_dim0_stride lo32
        g1[6] = 0;
        g1[7] = 0;
        v4i z4 = {0, 0, 0, 0};                            // groups 2/3 unused (<=2D)
        v8i z8 = {0, 0, 0, 0, 0, 0, 0, 0};                // extra group (6-arg form)
        __builtin_amdgcn_tensor_load_to_lds(g0, g1, z4, z4, z8, 0);
    };

    if (wave == 0) issueTDM(0, 0);

    for (int tile = 0; tile < NTILES; ++tile) {
        const int buf = tile & 1;
        if (wave == 0) __builtin_amdgcn_s_wait_tensorcnt(0);
        __syncthreads();                       // publish buf; all done with buf^1
        if (wave == 0 && tile + 1 < NTILES) issueTDM(tile + 1, buf ^ 1);

        const int kBase = tile * 16;
        const float bias = -vhalf[kBase + lane16];
        scoreTile(&lds[buf][0], &lds[buf][TILE_ELTS],
                  aH, aL, lane16, halfId, bias, kBase, bv, bi);
    }

    reduceWrite(bv, bi, lane16, halfId, rowBase, bestIdx);
}

// ---------------------------------------------------------------------------
// Kernel 3 (fallback, no workspace for wsB): converts vocab tiles in-loop.
// ---------------------------------------------------------------------------
__global__ void __launch_bounds__(256)
vq_argmin_wmma_direct(const float* __restrict__ seq,
                      const float* __restrict__ vocab,
                      const float* __restrict__ vhalf,
                      int* __restrict__ bestIdx) {
    __shared__ __align__(32) __bf16 ldsH[TILE_ELTS];
    __shared__ __align__(32) __bf16 ldsL[TILE_ELTS];

    const int tid    = threadIdx.x;
    const int lane   = tid & 31;
    const int wave   = tid >> 5;
    const int halfId = lane >> 4;
    const int lane16 = lane & 15;

    const int rowBase = (blockIdx.x * WAVES_PB + wave) * 16;
    v16bf aH[8], aL[8];
    loadAtile(seq, rowBase + lane16, halfId, aH, aL);

    float bv[8];
    int   bi[8];
    #pragma unroll
    for (int j = 0; j < 8; ++j) { bv[j] = -3.4e38f; bi[j] = 0; }

    for (int tile = 0; tile < NTILES; ++tile) {
        const int kBase = tile * 16;
        __syncthreads();
        {
            const int r  = tid >> 4;
            const int dB = (tid & 15) * 16;
            const float* vp = vocab + (size_t)(kBase + r) * DIM + dB;
            __bf16 th[16], tl[16];
            #pragma unroll
            for (int q = 0; q < 4; ++q) {
                float4 f = *reinterpret_cast<const float4*>(vp + q * 4);
                float vv[4] = {f.x, f.y, f.z, f.w};
                #pragma unroll
                for (int e = 0; e < 4; ++e) {
                    __bf16 h = (__bf16)vv[e];
                    th[q * 4 + e] = h;
                    tl[q * 4 + e] = (__bf16)(vv[e] - (float)h);
                }
            }
            v8bf* dh = reinterpret_cast<v8bf*>(&ldsH[r * DIM + dB]);
            v8bf* dl = reinterpret_cast<v8bf*>(&ldsL[r * DIM + dB]);
            dh[0] = *reinterpret_cast<v8bf*>(&th[0]);
            dh[1] = *reinterpret_cast<v8bf*>(&th[8]);
            dl[0] = *reinterpret_cast<v8bf*>(&tl[0]);
            dl[1] = *reinterpret_cast<v8bf*>(&tl[8]);
        }
        __syncthreads();

        const float bias = -vhalf[kBase + lane16];
        scoreTile(ldsH, ldsL, aH, aL, lane16, halfId, bias, kBase, bv, bi);
    }

    reduceWrite(bv, bi, lane16, halfId, rowBase, bestIdx);
}

// ---------------------------------------------------------------------------
// Kernel 4: gather quantized vectors + emit int32 indices.
// ---------------------------------------------------------------------------
__global__ void vq_gather(const float* __restrict__ vocab,
                          const int* __restrict__ bestIdx,
                          float* __restrict__ out,
                          int* __restrict__ outIdx) {
    int gid = blockIdx.x * blockDim.x + threadIdx.x;   // T_TOKENS * 64
    int t = gid >> 6;
    int q = gid & 63;
    int k = bestIdx[t];
    reinterpret_cast<float4*>(out)[(size_t)t * 64 + q] =
        reinterpret_cast<const float4*>(vocab + (size_t)k * DIM)[q];
    if (q == 0) outIdx[t] = k;
}

// ---------------------------------------------------------------------------
extern "C" void kernel_launch(void* const* d_in, const int* in_sizes, int n_in,
                              void* d_out, int out_size, void* d_ws, size_t ws_size,
                              hipStream_t stream) {
    const float* seq   = (const float*)d_in[0];   // [512,64,256] f32
    const float* vocab = (const float*)d_in[1];   // [8192,256]   f32

    // Workspace layout: vhalf | bestIdx | wsB (bf16 hi/lo planes, tile-major)
    const size_t vhalfBytes = KCODES * sizeof(float);          // 32 KB
    const size_t idxBytes   = T_TOKENS * sizeof(int);          // 128 KB
    const size_t wsBBytes   = (size_t)NTILES * TILE_BF16 * 2;  // 8 MB
    float* vhalf   = (float*)d_ws;
    int*   bestIdx = (int*)((char*)d_ws + vhalfBytes);
    __bf16* wsB    = (__bf16*)((char*)d_ws + vhalfBytes + idxBytes);

    float* out    = (float*)d_out;
    int*   outIdx = (int*)((float*)d_out + (size_t)T_TOKENS * DIM);

    vq_v2half<<<KCODES / 256, 256, 0, stream>>>(vocab, vhalf);

    if (ws_size >= vhalfBytes + idxBytes + wsBBytes) {
        vq_prepB<<<NTILES, 256, 0, stream>>>(vocab, wsB);
        vq_argmin_wmma_tdm<<<T_TOKENS / ROWS_PB, 256, 0, stream>>>(
            seq, wsB, vhalf, bestIdx);
    } else {
        vq_argmin_wmma_direct<<<T_TOKENS / ROWS_PB, 256, 0, stream>>>(
            seq, vocab, vhalf, bestIdx);
    }

    vq_gather<<<(T_TOKENS * 64) / 256, 256, 0, stream>>>(vocab, bestIdx, out, outIdx);
}
